// DependencyParsing_42296837931282
// MI455X (gfx1250) — compile-verified, matrix operand order checked by reading, not run on hardware
//
#include <hip/hip_runtime.h>
#include <hip/hip_bf16.h>

typedef __attribute__((ext_vector_type(16))) _Float16 v16h;
typedef __attribute__((ext_vector_type(8)))  _Float16 v8h;
typedef __attribute__((ext_vector_type(8)))  float    v8f;

#define B_TOTAL   65536
#define T_TOK     7
#define D_EMB     100
#define H_DIM     700
#define OUT_DIM   93
#define K1_REAL   2100
#define K1_TILES  66            // 2112 / 32
#define N1_TILES  44            // 704 / 16
#define N1_PAD    704
#define K2_TILES  22            // 704 / 32
#define N2_TILES  6             // 96 / 16
#define M_TILE    32
#define HS_STRIDE 720           // padded half-stride for h staging

#define WKP_HALVES (K1_TILES * N1_TILES * 512)   // 1,486,848 halves
#define WOP_HALVES (K2_TILES * N2_TILES * 512)   //    67,584 halves
#define OFF_WOP   (WKP_HALVES * 2)               // byte offsets in d_ws
#define OFF_BSUM  (OFF_WOP + WOP_HALVES * 2)
#define OFF_BO    (OFF_BSUM + N1_PAD * 4)

// ---------------------------------------------------------------------------
// Prep: pack stacked [Ww;Wp;Wd] (2112x704, zero padded) and Wo (704x96) into
// per-lane WMMA B-fragment order; fuse biases bw+bp+bd.
// Fragment order: element ((kt*NT + nt)*32 + lane)*16 + j holds
//   W[kt*32 + 8*(lane>>4) + (j&7) + ((j&8)?16:0)][nt*16 + (lane&15)]
// ---------------------------------------------------------------------------
__global__ __launch_bounds__(256)
void dep_parse_prep(const float* __restrict__ Ww, const float* __restrict__ Wp,
                    const float* __restrict__ Wd, const float* __restrict__ bw,
                    const float* __restrict__ bp, const float* __restrict__ bd,
                    const float* __restrict__ Wo, const float* __restrict__ bo,
                    _Float16* __restrict__ WKp, _Float16* __restrict__ WoP,
                    float* __restrict__ bsum, float* __restrict__ bopad)
{
    int i = blockIdx.x * 256 + threadIdx.x;
    if (i < WKP_HALVES) {
        int j = i & 15, l = (i >> 4) & 31, tile = i >> 9;
        int kt = tile / N1_TILES, nt = tile % N1_TILES;
        int k = kt * 32 + 8 * (l >> 4) + (j & 7) + ((j & 8) ? 16 : 0);
        int n = nt * 16 + (l & 15);
        float v = 0.0f;
        if (n < H_DIM && k < K1_REAL) {
            if (k < 700)       v = Ww[k * H_DIM + n];
            else if (k < 1400) v = Wp[(k - 700) * H_DIM + n];
            else               v = Wd[(k - 1400) * H_DIM + n];
        }
        WKp[i] = (_Float16)v;
    } else if (i < WKP_HALVES + WOP_HALVES) {
        int q = i - WKP_HALVES;
        int j = q & 15, l = (q >> 4) & 31, tile = q >> 9;
        int kt = tile / N2_TILES, nt = tile % N2_TILES;
        int k = kt * 32 + 8 * (l >> 4) + (j & 7) + ((j & 8) ? 16 : 0);
        int n = nt * 16 + (l & 15);
        float v = (k < H_DIM && n < OUT_DIM) ? Wo[k * OUT_DIM + n] : 0.0f;
        WoP[q] = (_Float16)v;
    } else if (i < WKP_HALVES + WOP_HALVES + N1_PAD) {
        int n = i - WKP_HALVES - WOP_HALVES;
        bsum[n] = (n < H_DIM) ? (bw[n] + bp[n] + bd[n]) : 0.0f;
    } else if (i < WKP_HALVES + WOP_HALVES + N1_PAD + 96) {
        int n = i - WKP_HALVES - WOP_HALVES - N1_PAD;
        bopad[n] = (n < OUT_DIM) ? bo[n] : 0.0f;
    }
}

static __device__ __forceinline__ v8f vzero8()
{
    v8f z = {0.f, 0.f, 0.f, 0.f, 0.f, 0.f, 0.f, 0.f};
    return z;
}

static __device__ __forceinline__ v16h make_a_frag(const _Float16* p)
{
    v8h lo = *(const v8h*)(p);
    v8h hi = *(const v8h*)(p + 16);
    return __builtin_shufflevector(lo, hi, 0, 1, 2, 3, 4, 5, 6, 7,
                                          8, 9, 10, 11, 12, 13, 14, 15);
}

// ---------------------------------------------------------------------------
// Main fused kernel: gather -> GEMM1 (K=2112) -> +bias, cube -> GEMM2 -> +bo
// -> softmax(93). One block = 32 rows, 8 wave32s.
// Wave layout GEMM1: mg = wave>>2 (M subtile), ng = wave&3 (11 N tiles each).
// ---------------------------------------------------------------------------
__global__ __launch_bounds__(256)
void dep_parse_main(const int* __restrict__ word_idx, const int* __restrict__ pos_idx,
                    const int* __restrict__ dep_idx,
                    const float* __restrict__ word_table,
                    const float* __restrict__ pos_table,
                    const float* __restrict__ dep_table,
                    const _Float16* __restrict__ WKp, const _Float16* __restrict__ WoP,
                    const float* __restrict__ bsum, const float* __restrict__ bopad,
                    float* __restrict__ out)
{
    __shared__ _Float16 Xs[M_TILE * 32];         //  2 KB gather staging
    __shared__ _Float16 Hs[M_TILE * HS_STRIDE];  // 45 KB h (cubed) staging
    __shared__ float    Ls[M_TILE * 96];         // 12 KB logits

    const int tid  = threadIdx.x;
    const int lane = tid & 31;
    const int wave = tid >> 5;
    const int mg   = wave >> 2;   // 0..1
    const int ng   = wave & 3;    // 0..3
    const int lhi  = lane >> 4;
    const int llo  = lane & 15;
    const int row0 = blockIdx.x * M_TILE;

    const v16h* WKv = (const v16h*)WKp;
    const v16h* WoV = (const v16h*)WoP;

    v8f c[11];
#pragma unroll
    for (int i = 0; i < 11; ++i) c[i] = vzero8();

    // ---- GEMM1: h = [we|pe|de] @ [Ww;Wp;Wd] ------------------------------
    for (int kt = 0; kt < K1_TILES; ++kt) {
        __syncthreads();  // previous iteration's A-frag reads are complete
        // cooperative gather of X[32 rows][32 k] into LDS as f16
#pragma unroll
        for (int e = 0; e < 4; ++e) {
            int p  = tid * 4 + e;
            int r  = p >> 5;
            int k  = kt * 32 + (p & 31);
            float v = 0.0f;
            if (k < K1_REAL) {
                int g    = k / 700;          // 0: word, 1: pos, 2: dep
                int k2   = k - g * 700;
                int t    = k2 / 100;
                int d    = k2 - t * 100;
                int rowg = (row0 + r) * T_TOK + t;
                if (g == 0) {
                    int wi = word_idx[rowg];
                    v = (wi >= 0) ? word_table[wi * D_EMB + d] : 0.0f;
                } else if (g == 1) {
                    v = pos_table[pos_idx[rowg] * D_EMB + d];
                } else {
                    v = dep_table[dep_idx[rowg] * D_EMB + d];
                }
            }
            Xs[p] = (_Float16)v;
        }
        __syncthreads();

        // A fragment from LDS (ISA 16-bit A layout: two contiguous 16B runs)
        int abase = (mg * 16 + llo) * 32 + 8 * lhi;
        v16h a = make_a_frag(&Xs[abase]);

        if (kt + 1 < K1_TILES)
            __builtin_prefetch(&WKv[((kt + 1) * N1_TILES + ng * 11) * 32 + lane], 0, 0);

#pragma unroll
        for (int i = 0; i < 11; ++i) {
            int nt = ng * 11 + i;
            v16h b = WKv[(kt * N1_TILES + nt) * 32 + lane];
            c[i] = __builtin_amdgcn_wmma_f32_16x16x32_f16(
                       false, a, false, b, (short)0, c[i], false, false);
        }
    }

    // ---- bias + cube, stage as f16 in LDS --------------------------------
#pragma unroll
    for (int i = 0; i < 11; ++i) {
        int n  = (ng * 11 + i) * 16 + llo;
        float bb = bsum[n];
#pragma unroll
        for (int r = 0; r < 8; ++r) {
            int m   = mg * 16 + 8 * lhi + r;   // C/D layout: M = r + 8*(lane>>4)
            float h = c[i][r] + bb;
            Hs[m * HS_STRIDE + n] = (_Float16)(h * h * h);
        }
    }
    __syncthreads();

    // ---- GEMM2: logits = h^3 @ Wo ---------------------------------------
    // tiles {ng} and {ng+4 if < 6}; uniform per wave so EXEC stays all-ones
    v8f c2a = vzero8(), c2b = vzero8();
    for (int kt = 0; kt < K2_TILES; ++kt) {
        int ab = (mg * 16 + llo) * HS_STRIDE + kt * 32 + 8 * lhi;
        v16h a = make_a_frag(&Hs[ab]);
        v16h b0 = WoV[(kt * N2_TILES + ng) * 32 + lane];
        c2a = __builtin_amdgcn_wmma_f32_16x16x32_f16(
                  false, a, false, b0, (short)0, c2a, false, false);
        if (ng < 2) {
            v16h b1 = WoV[(kt * N2_TILES + ng + 4) * 32 + lane];
            c2b = __builtin_amdgcn_wmma_f32_16x16x32_f16(
                      false, a, false, b1, (short)0, c2b, false, false);
        }
    }

    {
        int n = ng * 16 + llo;
        float bb = bopad[n];
#pragma unroll
        for (int r = 0; r < 8; ++r)
            Ls[(mg * 16 + 8 * lhi + r) * 96 + n] = c2a[r] + bb;
        if (ng < 2) {
            int n2 = (ng + 4) * 16 + llo;
            float bb2 = bopad[n2];
#pragma unroll
            for (int r = 0; r < 8; ++r)
                Ls[(mg * 16 + 8 * lhi + r) * 96 + n2] = c2b[r] + bb2;
        }
    }
    __syncthreads();

    // ---- softmax over 93 classes, one row per thread ---------------------
    if (tid < M_TILE) {
        const float* lrow = &Ls[tid * 96];
        float mx = -1e30f;
        for (int j = 0; j < OUT_DIM; ++j) mx = fmaxf(mx, lrow[j]);
        float s = 0.0f;
        for (int j = 0; j < OUT_DIM; ++j) s += __expf(lrow[j] - mx);
        float inv = 1.0f / s;
        float* orow = &out[(size_t)(row0 + tid) * OUT_DIM];
        for (int j = 0; j < OUT_DIM; ++j)
            orow[j] = __expf(lrow[j] - mx) * inv;
    }
}

extern "C" void kernel_launch(void* const* d_in, const int* in_sizes, int n_in,
                              void* d_out, int out_size, void* d_ws, size_t ws_size,
                              hipStream_t stream) {
    const int*   word_idx   = (const int*)d_in[0];
    const int*   pos_idx    = (const int*)d_in[1];
    const int*   dep_idx    = (const int*)d_in[2];
    const float* word_table = (const float*)d_in[3];
    const float* pos_table  = (const float*)d_in[4];
    const float* dep_table  = (const float*)d_in[5];
    const float* Ww = (const float*)d_in[6];
    const float* bw = (const float*)d_in[7];
    const float* Wp = (const float*)d_in[8];
    const float* bp = (const float*)d_in[9];
    const float* Wd = (const float*)d_in[10];
    const float* bd = (const float*)d_in[11];
    const float* Wo = (const float*)d_in[12];
    const float* bo = (const float*)d_in[13];

    _Float16* WKp  = (_Float16*)d_ws;
    _Float16* WoP  = (_Float16*)((char*)d_ws + OFF_WOP);
    float*    bsum = (float*)((char*)d_ws + OFF_BSUM);
    float*    bopad= (float*)((char*)d_ws + OFF_BO);

    const int prep_elems = WKP_HALVES + WOP_HALVES + N1_PAD + 96;
    dep_parse_prep<<<(prep_elems + 255) / 256, 256, 0, stream>>>(
        Ww, Wp, Wd, bw, bp, bd, Wo, bo, WKp, WoP, bsum, bopad);

    dep_parse_main<<<B_TOTAL / M_TILE, 256, 0, stream>>>(
        word_idx, pos_idx, dep_idx, word_table, pos_table, dep_table,
        WKp, WoP, bsum, bopad, (float*)d_out);
}
